// GDN_12695923327310
// MI455X (gfx1250) — compile-verified, matrix operand order checked by reading, not run on hardware
//
#include <hip/hip_runtime.h>
#include <hip/hip_bf16.h>

#define MTOT  32768   // B*N
#define NN    512
#define BB    64
#define DD    128
#define KTOP  32
#define FIN   128
#define INTERD 256
#define BN_EPS 1e-5f

typedef __attribute__((ext_vector_type(16))) _Float16 v16h;
typedef __attribute__((ext_vector_type(8)))  _Float16 v8h;
typedef __attribute__((ext_vector_type(8)))  float    v8f;

// 16-byte int vector for the async-to-LDS builtin (type from hipcc diagnostic)
typedef int v4i_vec __attribute__((vector_size(16)));
typedef __attribute__((address_space(1))) v4i_vec gbl_v4i;
typedef __attribute__((address_space(3))) v4i_vec lds_v4i;

#if defined(__has_builtin)
#if __has_builtin(__builtin_amdgcn_global_load_async_to_lds_b128) && \
    __has_builtin(__builtin_amdgcn_s_wait_asynccnt)
#define HAS_ASYNC_LDS 1
#endif
#endif
#ifndef HAS_ASYNC_LDS
#define HAS_ASYNC_LDS 0
#endif

// ---------------------------------------------------------------------------
// 1) cosine similarity + top-32 via stable rank (matches jax.lax.top_k order)
// ---------------------------------------------------------------------------
__global__ __launch_bounds__(512) void k_topk(const float* __restrict__ emb,
                                              int* __restrict__ topk) {
    const int i = blockIdx.x;    // target row
    const int j = threadIdx.x;   // candidate column
    __shared__ float wi[DD];
    __shared__ float cosv[NN];
    if (j < DD) wi[j] = emb[i * DD + j];
    __syncthreads();
    float dot = 0.f, ni = 0.f, nj = 0.f;
    for (int d = 0; d < DD; ++d) {
        float a = wi[d];
        float b = emb[j * DD + d];
        dot += a * b; ni += a * a; nj += b * b;
    }
    float c = dot * rsqrtf(ni * nj);
    cosv[j] = c;
    __syncthreads();
    int rank = 0;
    for (int m = 0; m < NN; ++m) {
        float cm = cosv[m];
        rank += (cm > c) || (cm == c && m < j);
    }
    if (rank < KTOP) topk[i * KTOP + rank] = j;
}

// ---------------------------------------------------------------------------
// 2a) one-time weight prep: W[K][COUT] f32  ->  Wt[COUT][K] f16 (transposed)
// ---------------------------------------------------------------------------
__global__ void k_prep_w(const float* __restrict__ W, _Float16* __restrict__ Wt,
                         int K, int COUT) {
    const int idx = blockIdx.x * 256 + threadIdx.x;  // over K*COUT, coalesced read
    const int k = idx / COUT, n = idx % COUT;
    Wt[(size_t)n * K + k] = (_Float16)W[idx];
}

// ---------------------------------------------------------------------------
// 2b) fp32-in/fp32-out GEMM via v_wmma_f32_16x16x32_f16.
//     One wave -> 16 rows x 64 cols of C (4 accumulators): the A fragment is
//     loaded/converted once per K-step and reused by 4 WMMAs.
//     Pre-transposed f16 W panel staged in LDS each K-step; when available,
//     staging uses GLOBAL_LOAD_ASYNC_TO_LDS_B128 (ASYNCcnt path, no VGPRs).
// ---------------------------------------------------------------------------
template <int KDIM, int COUT>
__global__ __launch_bounds__(256) void k_gemm(const float* __restrict__ A,
                                              const _Float16* __restrict__ Wt,
                                              float* __restrict__ C) {
    __shared__ __align__(16) _Float16 sW[COUT * 32];  // sW[n*32 + kk]
    const int wave   = threadIdx.x >> 5;
    const int lane   = threadIdx.x & 31;
    const int tilesN4 = COUT / 64;                // 64-wide column groups
    const int tile   = blockIdx.x * 8 + wave;
    const int tm     = tile / tilesN4;
    const int tn4    = tile % tilesN4;
    const int l16    = lane & 15;
    const int khalf  = lane >> 4;                 // 0: K{0-7,16-23}, 1: K{8-15,24-31}
    const int row    = tm * 16 + l16;             // A row for this lane
    const int ncol0  = tn4 * 64 + l16;            // first C column for this lane

    v8f acc0 = {}, acc1 = {}, acc2 = {}, acc3 = {};
    for (int k0 = 0; k0 < KDIM; k0 += 32) {
        __syncthreads();
        // stage Wt[n][k0..k0+31] -> sW[n*32 .. +31] as 16-byte chunks
        const int chunks = COUT * 4;              // 16B chunks in the panel
        for (int c = threadIdx.x; c < chunks; c += 256) {
            const int n = c >> 2, part = c & 3;
            const _Float16* gp = Wt + (size_t)n * KDIM + k0 + part * 8;
            _Float16*       lp = sW + n * 32 + part * 8;
#if HAS_ASYNC_LDS
            __builtin_amdgcn_global_load_async_to_lds_b128(
                (gbl_v4i*)gp, (lds_v4i*)lp, 0, 0);
#else
            *(v8h*)lp = *(const v8h*)gp;
#endif
        }
#if HAS_ASYNC_LDS
        __builtin_amdgcn_s_wait_asynccnt(0);
#endif
        __syncthreads();

        // A fragment: 16x32 f16 layout (loaded once, reused by 4 WMMAs)
        v16h af;
        const float* ap = A + (size_t)row * KDIM + k0 + khalf * 8;
        #pragma unroll
        for (int j = 0; j < 8; ++j) af[j]     = (_Float16)ap[j];      // K = khalf*8 + j
        #pragma unroll
        for (int j = 0; j < 8; ++j) af[8 + j] = (_Float16)ap[16 + j]; // K = 16 + khalf*8 + j

        // prefetch next K panel of A toward the WGP while WMMAs run
        if (k0 + 32 < KDIM) __builtin_prefetch(ap + 32, 0, 3);

        // 4 B fragments (columns ncol0, +16, +32, +48), two 16B LDS loads each
        #pragma unroll
        for (int sub = 0; sub < 4; ++sub) {
            const int ncol = ncol0 + sub * 16;
            v8h b0 = *(const v8h*)(sW + ncol * 32 + khalf * 8);
            v8h b1 = *(const v8h*)(sW + ncol * 32 + 16 + khalf * 8);
            v16h bf;
            #pragma unroll
            for (int j = 0; j < 8; ++j) { bf[j] = b0[j]; bf[8 + j] = b1[j]; }
            if (sub == 0)
                acc0 = __builtin_amdgcn_wmma_f32_16x16x32_f16(false, af, false, bf, (short)0, acc0, false, false);
            else if (sub == 1)
                acc1 = __builtin_amdgcn_wmma_f32_16x16x32_f16(false, af, false, bf, (short)0, acc1, false, false);
            else if (sub == 2)
                acc2 = __builtin_amdgcn_wmma_f32_16x16x32_f16(false, af, false, bf, (short)0, acc2, false, false);
            else
                acc3 = __builtin_amdgcn_wmma_f32_16x16x32_f16(false, af, false, bf, (short)0, acc3, false, false);
        }
    }
    // C/D layout: lane<16 -> rows r, lane>=16 -> rows r+8
    float* cp = C + (size_t)(tm * 16 + khalf * 8) * COUT + ncol0;
    #pragma unroll
    for (int r = 0; r < 8; ++r) {
        cp[(size_t)r * COUT]      = acc0[r];
        cp[(size_t)r * COUT + 16] = acc1[r];
        cp[(size_t)r * COUT + 32] = acc2[r];
        cp[(size_t)r * COUT + 48] = acc3[r];
    }
}

// ---------------------------------------------------------------------------
// 3) per-node attention scores: s_i = x_lin.att_i + emb.att_em_i (one wave/node)
// ---------------------------------------------------------------------------
__global__ __launch_bounds__(256) void k_scores(const float* __restrict__ xlin,
                                                const float* __restrict__ emb,
                                                const float* __restrict__ att_i,
                                                const float* __restrict__ att_j,
                                                const float* __restrict__ att_em_i,
                                                const float* __restrict__ att_em_j,
                                                float* __restrict__ s_i,
                                                float* __restrict__ s_j) {
    const int wave = threadIdx.x >> 5, lane = threadIdx.x & 31;
    const int m = blockIdx.x * 8 + wave;
    const int n = m & (NN - 1);
    const float4 xl  = *(const float4*)(xlin + (size_t)m * DD + lane * 4);
    const float4 eb  = *(const float4*)(emb  + (size_t)n * DD + lane * 4);
    const float4 ai  = *(const float4*)(att_i    + lane * 4);
    const float4 aj  = *(const float4*)(att_j    + lane * 4);
    const float4 aei = *(const float4*)(att_em_i + lane * 4);
    const float4 aej = *(const float4*)(att_em_j + lane * 4);
    float pi = xl.x*ai.x + xl.y*ai.y + xl.z*ai.z + xl.w*ai.w
             + eb.x*aei.x + eb.y*aei.y + eb.z*aei.z + eb.w*aei.w;
    float pj = xl.x*aj.x + xl.y*aj.y + xl.z*aj.z + xl.w*aj.w
             + eb.x*aej.x + eb.y*aej.y + eb.z*aej.z + eb.w*aej.w;
    for (int off = 16; off; off >>= 1) {
        pi += __shfl_xor(pi, off, 32);
        pj += __shfl_xor(pj, off, 32);
    }
    if (lane == 0) { s_i[m] = pi; s_j[m] = pj; }
}

// ---------------------------------------------------------------------------
// 4) edge softmax + aggregation (one wave per destination node; lane = edge)
// ---------------------------------------------------------------------------
__global__ __launch_bounds__(256) void k_attn(const float* __restrict__ xlin,
                                              const int* __restrict__ topk,
                                              const float* __restrict__ s_i,
                                              const float* __restrict__ s_j,
                                              float* __restrict__ agg) {
    const int wave = threadIdx.x >> 5, lane = threadIdx.x & 31;
    const int m = blockIdx.x * 8 + wave;
    const int b = m >> 9;            // m / NN
    const int i = m & (NN - 1);
    const float si = s_i[m];

    const int src_n = topk[i * KTOP + lane];
    const int src_m = b * NN + src_n;
    const bool valid = (src_n != i);

    float raw   = si + s_j[src_m];
    float logit = raw > 0.f ? raw : 0.2f * raw;       // leaky_relu(.,0.2)
    float lg    = valid ? logit : -1e9f;

    float rself = si + s_j[m];
    float lself = rself > 0.f ? rself : 0.2f * rself; // self loop always valid

    float mx = lg;
    for (int off = 16; off; off >>= 1) mx = fmaxf(mx, __shfl_xor(mx, off, 32));
    mx = fmaxf(mx, lself);

    float e     = valid ? __expf(logit - mx) : 0.0f;
    float eself = __expf(lself - mx);
    float den = e;
    for (int off = 16; off; off >>= 1) den += __shfl_xor(den, off, 32);
    den += eself;
    const float alpha  = e / den;
    const float aself  = eself / den;

    float4 acc = {0.f, 0.f, 0.f, 0.f};
    for (int k = 0; k < KTOP; ++k) {
        float a = __shfl(alpha, k, 32);
        int   sm = __shfl(src_m, k, 32);
        if (a != 0.0f) {
            float4 v = *(const float4*)(xlin + (size_t)sm * DD + lane * 4);
            acc.x += a * v.x; acc.y += a * v.y; acc.z += a * v.z; acc.w += a * v.w;
        }
    }
    float4 v = *(const float4*)(xlin + (size_t)m * DD + lane * 4);
    acc.x += aself * v.x; acc.y += aself * v.y; acc.z += aself * v.z; acc.w += aself * v.w;
    *(float4*)(agg + (size_t)m * DD + lane * 4) = acc;
}

// ---------------------------------------------------------------------------
// 5) BatchNorm: two-stage mean/var over the M axis. Pre-BN per-channel biases
//    cancel in training-mode BN, so they are dropped entirely.
// ---------------------------------------------------------------------------
template <int C>
__global__ void k_bnstats(const float* __restrict__ x, float* __restrict__ partial) {
    const int t = threadIdx.x;                 // channel
    const int rows = MTOT / 128;               // 128 blocks
    const int r0 = blockIdx.x * rows;
    float s = 0.f, q = 0.f;
    for (int r = 0; r < rows; ++r) {
        float v = x[(size_t)(r0 + r) * C + t];
        s += v; q += v * v;
    }
    partial[(size_t)blockIdx.x * 2 * C + t]     = s;
    partial[(size_t)blockIdx.x * 2 * C + C + t] = q;
}

template <int C>
__global__ void k_bnfin(const float* __restrict__ partial,
                        const float* __restrict__ g, const float* __restrict__ bnb,
                        float* __restrict__ scale, float* __restrict__ shift) {
    const int t = threadIdx.x;
    float s = 0.f, q = 0.f;
    for (int b = 0; b < 128; ++b) {
        s += partial[(size_t)b * 2 * C + t];
        q += partial[(size_t)b * 2 * C + C + t];
    }
    const float mean = s * (1.0f / (float)MTOT);
    const float var  = q * (1.0f / (float)MTOT) - mean * mean;
    const float sc   = g[t] * rsqrtf(var + BN_EPS);
    scale[t] = sc;
    shift[t] = bnb[t] - mean * sc;
}

// gcn = relu(bn(agg)); h = gcn * emb  (gating)
__global__ void k_bn_relu_gate(const float* __restrict__ x,
                               const float* __restrict__ scale,
                               const float* __restrict__ shift,
                               const float* __restrict__ emb,
                               float* __restrict__ h) {
    const int idx = blockIdx.x * 256 + threadIdx.x;     // over MTOT*DD
    const int c = idx & (DD - 1);
    const int m = idx >> 7;
    const int n = m & (NN - 1);
    float v = x[idx] * scale[c] + shift[c];
    v = v > 0.f ? v : 0.f;
    h[idx] = v * emb[n * DD + c];
}

template <int C>
__global__ void k_bn_relu(const float* __restrict__ x,
                          const float* __restrict__ scale,
                          const float* __restrict__ shift,
                          float* __restrict__ y) {
    const int idx = blockIdx.x * 256 + threadIdx.x;
    const int c = idx & (C - 1);
    float v = x[idx] * scale[c] + shift[c];
    y[idx] = v > 0.f ? v : 0.f;
}

// ---------------------------------------------------------------------------
// 6) final projection: out[m] = act2[m,:] . W3 + b3   (one wave per node)
// ---------------------------------------------------------------------------
__global__ __launch_bounds__(256) void k_final(const float* __restrict__ x,
                                               const float* __restrict__ W3,
                                               const float* __restrict__ b3,
                                               float* __restrict__ out) {
    const int wave = threadIdx.x >> 5, lane = threadIdx.x & 31;
    const int m = blockIdx.x * 8 + wave;
    float p = 0.f;
    #pragma unroll
    for (int j = 0; j < 8; ++j)
        p += x[(size_t)m * INTERD + lane + j * 32] * W3[lane + j * 32];
    for (int off = 16; off; off >>= 1) p += __shfl_xor(p, off, 32);
    if (lane == 0) out[m] = p + b3[0];
}

// ---------------------------------------------------------------------------
extern "C" void kernel_launch(void* const* d_in, const int* in_sizes, int n_in,
                              void* d_out, int out_size, void* d_ws, size_t ws_size,
                              hipStream_t stream) {
    const float* data     = (const float*)d_in[0];   // [M, FIN]
    const float* emb      = (const float*)d_in[1];   // [N, D]
    const float* lin_W    = (const float*)d_in[2];   // [FIN, D]
    const float* att_i    = (const float*)d_in[3];
    const float* att_j    = (const float*)d_in[4];
    const float* att_em_i = (const float*)d_in[5];
    const float* att_em_j = (const float*)d_in[6];
    // d_in[7] = gnn_bias  (cancels in BN)
    const float* gnn_bn_g = (const float*)d_in[8];
    const float* gnn_bn_b = (const float*)d_in[9];
    const float* bn_out_g = (const float*)d_in[10];
    const float* bn_out_b = (const float*)d_in[11];
    const float* W1       = (const float*)d_in[12];  // [D, INTER]
    // d_in[13] = b1 (cancels)
    const float* bn1_g    = (const float*)d_in[14];
    const float* bn1_b    = (const float*)d_in[15];
    const float* W2       = (const float*)d_in[16];  // [INTER, INTER]
    // d_in[17] = b2 (cancels)
    const float* bn2_g    = (const float*)d_in[18];
    const float* bn2_b    = (const float*)d_in[19];
    const float* W3       = (const float*)d_in[20];  // [INTER, 1]
    const float* b3       = (const float*)d_in[21];

    float* w = (float*)d_ws;
    float* bufA    = w;                         // [M,128]  x_lin -> h -> (act2 lo)
    float* bufB    = bufA + (size_t)MTOT * DD;  // [M,128]  agg -> hact -> (act2 hi)
    float* bufC    = bufB + (size_t)MTOT * DD;  // [M,256]  act1
    float* s_i     = bufC + (size_t)MTOT * INTERD;
    float* s_j     = s_i + MTOT;
    float* scale   = s_j + MTOT;                // [256]
    float* shift   = scale + 256;               // [256]
    float* partial = shift + 256;               // 128 * 2 * 256
    _Float16* Wt   = (_Float16*)(partial + 128 * 2 * 256);  // [<=256][<=256] f16, 16B aligned
    int*   topk    = (int*)((float*)Wt + (INTERD * INTERD) / 2);
    float* act2    = bufA;                      // [M,256] spans bufA+bufB

    const int EW = MTOT / 8;                   // wave-per-node grids

    // graph construction
    k_topk<<<NN, 512, 0, stream>>>(emb, topk);

    // x_lin = data @ lin_W     (WMMA, 16x64 per wave)
    k_prep_w<<<(FIN * DD) / 256, 256, 0, stream>>>(lin_W, Wt, FIN, DD);
    k_gemm<FIN, DD><<<(MTOT / 16) * (DD / 64) / 8, 256, 0, stream>>>(data, Wt, bufA);

    // attention scores, softmax, aggregation
    k_scores<<<EW, 256, 0, stream>>>(bufA, emb, att_i, att_j, att_em_i, att_em_j, s_i, s_j);
    k_attn<<<EW, 256, 0, stream>>>(bufA, topk, s_i, s_j, bufB);

    // gcn = relu(bn(agg)); h = gcn * emb
    k_bnstats<DD><<<128, DD, 0, stream>>>(bufB, partial);
    k_bnfin<DD><<<1, DD, 0, stream>>>(partial, gnn_bn_g, gnn_bn_b, scale, shift);
    k_bn_relu_gate<<<(MTOT * DD) / 256, 256, 0, stream>>>(bufB, scale, shift, emb, bufA);

    // h = relu(bn(h))
    k_bnstats<DD><<<128, DD, 0, stream>>>(bufA, partial);
    k_bnfin<DD><<<1, DD, 0, stream>>>(partial, bn_out_g, bn_out_b, scale, shift);
    k_bn_relu<DD><<<(MTOT * DD) / 256, 256, 0, stream>>>(bufA, scale, shift, bufB);

    // o1 = relu(bn(h @ W1))    (WMMA)
    k_prep_w<<<(DD * INTERD) / 256, 256, 0, stream>>>(W1, Wt, DD, INTERD);
    k_gemm<DD, INTERD><<<(MTOT / 16) * (INTERD / 64) / 8, 256, 0, stream>>>(bufB, Wt, bufC);
    k_bnstats<INTERD><<<128, INTERD, 0, stream>>>(bufC, partial);
    k_bnfin<INTERD><<<1, INTERD, 0, stream>>>(partial, bn1_g, bn1_b, scale, shift);
    k_bn_relu<INTERD><<<(MTOT * INTERD) / 256, 256, 0, stream>>>(bufC, scale, shift, bufC);

    // o2 = relu(bn(o1 @ W2))   (WMMA)
    k_prep_w<<<(INTERD * INTERD) / 256, 256, 0, stream>>>(W2, Wt, INTERD, INTERD);
    k_gemm<INTERD, INTERD><<<(MTOT / 16) * (INTERD / 64) / 8, 256, 0, stream>>>(bufC, Wt, act2);
    k_bnstats<INTERD><<<128, INTERD, 0, stream>>>(act2, partial);
    k_bnfin<INTERD><<<1, INTERD, 0, stream>>>(partial, bn2_g, bn2_b, scale, shift);
    k_bn_relu<INTERD><<<(MTOT * INTERD) / 256, 256, 0, stream>>>(act2, scale, shift, act2);

    // out = o2 @ W3 + b3
    k_final<<<EW, 256, 0, stream>>>(act2, W3, b3, (float*)d_out);
}
